// Projector_6244882448715
// MI455X (gfx1250) — matrix operand order repaired and checked by reading, not compiled, not run on hardware
//
#include <hip/hip_runtime.h>
#include <hip/hip_bf16.h>
#include <stdint.h>

typedef __attribute__((ext_vector_type(16))) _Float16 v16h;
typedef __attribute__((ext_vector_type(8)))  float    v8f;

#define NQ 12
#define BATCH 16
#define WDIM 512
#define ODIM 2305   // 256*9 + 1

// ---------------------------------------------------------------------------
// A-fragment K-local index for V_WMMA_F32_16X16X32_F16 (contiguous k = 16*(lane/16) + e, same order as the B fragment)
// ---------------------------------------------------------------------------
__device__ __forceinline__ int a_klocal(int lane, int e) {
    int j = e >> 1, lo = e & 1;
    int base;
    if (lane < 16) base = 2 * j;
    else           base = 16 + 2 * j;
    return base + lo;
}

// ---------------------------------------------------------------------------
// Pack OIHW f32 weights into fragment-ready f16 with permuted reduction order:
//   k' = (ky*KS + kx)*CIN + c   (channels innermost; CIN % 32 == 0)
// ---------------------------------------------------------------------------
template <int CIN, int KS>
__global__ void pack_w_kernel(const float* __restrict__ w, _Float16* __restrict__ wpk, int cout) {
    const int K   = CIN * KS * KS;
    const int NKC = K / 32;
    int idx = blockIdx.x * 256 + threadIdx.x;
    if (idx >= cout * K) return;
    int e    = idx & 15;
    int lane = (idx >> 4) & 31;
    int kc   = (idx >> 9) % NKC;
    int mt   = idx / (NKC * 512);
    int m    = mt * 16 + (lane & 15);
    int k    = kc * 32 + a_klocal(lane, e);   // permuted K index
    int c    = k % CIN;
    int kyx  = k / CIN;
    int ky   = kyx / KS, kx = kyx % KS;
    wpk[idx] = (_Float16)w[((size_t)(m * CIN + c) * KS + ky) * KS + kx];
}

// ---------------------------------------------------------------------------
// Bilinear 2x upsample (half-pixel centers), output NHWC f16.
// IN_NCHW selects input layout (first stage input is the NCHW f32 tensor).
// ---------------------------------------------------------------------------
template <typename Tin, bool IN_NCHW>
__global__ void up2_kernel(const Tin* __restrict__ in, _Float16* __restrict__ out,
                           int B, int C, int H, int W) {
    int OW = 2 * W, OH = 2 * H;
    long long idx = (long long)blockIdx.x * 256 + threadIdx.x;
    long long total = (long long)B * OH * OW * C;
    if (idx >= total) return;
    int c = (int)(idx % C);
    long long t = idx / C;
    int ox = (int)(t % OW); t /= OW;
    int oy = (int)(t % OH);
    int b  = (int)(t / OH);
    float sy = oy * 0.5f - 0.25f, sx = ox * 0.5f - 0.25f;
    int y0 = (int)floorf(sy), x0 = (int)floorf(sx);
    float fy = sy - (float)y0, fx = sx - (float)x0;
    int y0c = min(max(y0, 0), H - 1), y1c = min(max(y0 + 1, 0), H - 1);
    int x0c = min(max(x0, 0), W - 1), x1c = min(max(x0 + 1, 0), W - 1);
    float v00, v01, v10, v11;
    if (IN_NCHW) {
        const Tin* p = in + ((size_t)b * C + c) * H * W;
        v00 = (float)p[y0c * W + x0c]; v01 = (float)p[y0c * W + x1c];
        v10 = (float)p[y1c * W + x0c]; v11 = (float)p[y1c * W + x1c];
    } else {
        const Tin* p = in + (size_t)b * H * W * C + c;
        v00 = (float)p[((size_t)y0c * W + x0c) * C]; v01 = (float)p[((size_t)y0c * W + x1c) * C];
        v10 = (float)p[((size_t)y1c * W + x0c) * C]; v11 = (float)p[((size_t)y1c * W + x1c) * C];
    }
    float v = v00 * (1.f - fy) * (1.f - fx) + v01 * (1.f - fy) * fx +
              v10 * fy * (1.f - fx) + v11 * fy * fx;
    out[idx] = (_Float16)v;
}

// ---------------------------------------------------------------------------
// BN fold: st = [s1(512), t1(512), s2(256), t2(256), s3=1(256), t3=conv3_b(256)]
// ---------------------------------------------------------------------------
__global__ void prep_st_kernel(const float* g1, const float* b1, const float* m1, const float* v1,
                               const float* g2, const float* b2, const float* m2, const float* v2,
                               const float* c3b, float* st) {
    int i = blockIdx.x * 256 + threadIdx.x;
    if (i < 512) {
        float s = g1[i] * rsqrtf(v1[i] + 1e-5f);
        st[i] = s; st[512 + i] = b1[i] - m1[i] * s;
    } else if (i < 768) {
        int c = i - 512;
        float s = g2[c] * rsqrtf(v2[c] + 1e-5f);
        st[1024 + c] = s; st[1280 + c] = b2[c] - m2[c] * s;
    } else if (i < 1024) {
        int c = i - 768;
        st[1536 + c] = 1.0f; st[1792 + c] = c3b[c];
    }
}

// ---------------------------------------------------------------------------
// Implicit-GEMM conv with WMMA, NHWC activations. Block = 256 threads = 8 waves.
//  wave m-tile: 16 output channels (128 Cout per block);
//  N-tile: 16 contiguous x positions of one output row.
//  LDS tile [ky][tx][c] (c contiguous) filled with per-lane
//  GLOBAL_LOAD_ASYNC_TO_LDS_B128 (ASYNCcnt), zeros for the halo.
//  grid = (ceil(W/16), H, B * cout/128)
// ---------------------------------------------------------------------------
template <int CIN, int KS>
__global__ __launch_bounds__(256)
void conv_wmma_kernel(const _Float16* __restrict__ in, const _Float16* __restrict__ wpk,
                      const float* __restrict__ scale, const float* __restrict__ shift,
                      _Float16* __restrict__ out, int H, int W, int cout, int relu) {
    const int PAD = KS / 2;
    const int TW  = 16 + 2 * PAD;
    const int K   = CIN * KS * KS;
    const int NKC = K / 32;
    const int CPC = CIN / 32;     // K-chunks per (ky,kx) plane

    __shared__ alignas(32) _Float16 tile[KS * TW * CIN];   // [ky][tx][c]

    const int x0   = blockIdx.x * 16;
    const int y    = blockIdx.y;
    const int nblk = cout >> 7;
    const int cb   = blockIdx.z % nblk;
    const int b    = blockIdx.z / nblk;
    const int tid  = threadIdx.x;
    const int lane = tid & 31;
    const int wv   = tid >> 5;

    // --- stage NHWC halo tile into LDS with async 16B copies ---
    const _Float16* inb = in + (size_t)b * H * W * CIN;
    const int NV = KS * TW * (CIN / 8);          // 16-byte vectors
    for (int i = tid; i < NV; i += 256) {
        int cv = i % (CIN / 8);
        int t2 = i / (CIN / 8);                  // = ky*TW + tx
        int tx = t2 % TW;
        int ky = t2 / TW;
        int gy = y + ky - PAD;
        int gx = x0 + tx - PAD;
        _Float16* dst = &tile[(size_t)t2 * CIN + cv * 8];
        if (gy >= 0 && gy < H && gx >= 0 && gx < W) {
            const _Float16* gp = inb + (((size_t)gy * W + gx) * CIN + cv * 8);
            unsigned ldsoff = (unsigned)(uintptr_t)dst;
            asm volatile("global_load_async_to_lds_b128 %0, %1, off"
                         :: "v"(ldsoff), "v"(gp) : "memory");
        } else {
            uint4 z = make_uint4(0u, 0u, 0u, 0u);
            *(uint4*)dst = z;
        }
    }
    asm volatile("s_wait_asynccnt 0" ::: "memory");
    __syncthreads();

    // --- main WMMA loop over permuted K chunks ---
    const int mt = cb * 8 + wv;              // global m-tile
    const int n  = lane & 15;                // pixel index within N tile
    const int kb = (lane & 16) ? 16 : 0;     // B-fragment K base within chunk
    const _Float16* wbase = wpk + ((size_t)mt * NKC) * 512 + (size_t)lane * 16;

    v8f acc = {0.f, 0.f, 0.f, 0.f, 0.f, 0.f, 0.f, 0.f};
#pragma unroll 4
    for (int kc = 0; kc < NKC; ++kc) {
        v16h a = *(const v16h*)(wbase + (size_t)kc * 512);
        int kyx   = kc / CPC;                 // fixed (ky,kx) within chunk
        int cbase = (kc % CPC) * 32;
        int ky = kyx / KS, kx = kyx % KS;
        const _Float16* bp = tile + ((size_t)(ky * TW + n + kx)) * CIN + cbase + kb;
        v16h bv = *(const v16h*)bp;           // 16 contiguous halfs, 32B aligned
        acc = __builtin_amdgcn_wmma_f32_16x16x32_f16(false, a, false, bv,
                                                     (short)0, acc, false, false);
    }

    // --- fused BN(+ReLU) epilogue: 8 consecutive NHWC channels, one 16B store ---
    int x = x0 + n;
    if (x < W) {
        int mo = (lane & 16) ? 8 : 0;
        int ob = cb * 128 + wv * 16 + mo;     // first of 8 consecutive channels
        _Float16 tmp[8];
#pragma unroll
        for (int r = 0; r < 8; ++r) {
            float v = acc[r] * scale[ob + r] + shift[ob + r];
            if (relu) v = v > 0.f ? v : 0.f;
            tmp[r] = (_Float16)v;
        }
        _Float16* op = out + (((size_t)b * H + y) * W + x) * cout + ob;
        *(uint4*)op = *(uint4*)tmp;
    }
}

// ---------------------------------------------------------------------------
// Fold queries: wbar[b,d] = sum_q score[q,b]*word[q,b,d];  sbar[b] = sum_q score
// ---------------------------------------------------------------------------
__global__ void fold_word_kernel(const float* __restrict__ word, const float* __restrict__ score,
                                 float* __restrict__ wbar, float* __restrict__ sbar) {
    int i = blockIdx.x * 256 + threadIdx.x;
    if (i >= BATCH * WDIM) return;
    int d = i & (WDIM - 1);
    int b = i >> 9;
    float acc = 0.f;
    for (int q = 0; q < NQ; ++q)
        acc += score[q * BATCH + b] * word[((size_t)(q * BATCH + b)) * WDIM + d];
    wbar[i] = acc;
    if (d == 0) {
        float s = 0.f;
        for (int q = 0; q < NQ; ++q) s += score[q * BATCH + b];
        sbar[b] = s;
    }
}

// F[b,o] = txt_w[o,:].wbar[b,:] + sbar[b]*txt_b[o]   (o < 2305; o==2304 is bias)
__global__ void dyn_filter_kernel(const float* __restrict__ txt_w, const float* __restrict__ txt_b,
                                  const float* __restrict__ wbar, const float* __restrict__ sbar,
                                  float* __restrict__ F) {
    int i = blockIdx.x * 256 + threadIdx.x;
    if (i >= BATCH * ODIM) return;
    int o = i % ODIM;
    int b = i / ODIM;
    const float* wr = txt_w + (size_t)o * WDIM;
    const float* wb = wbar + (size_t)b * WDIM;
    float acc = 0.f;
    for (int d = 0; d < WDIM; ++d) acc += wr[d] * wb[d];
    F[i] = acc + sbar[b] * txt_b[o];
}

// ---------------------------------------------------------------------------
// Final folded dynamic conv over NHWC h:
//   out[b,y,x] = sum_{ky,kx} sum_c Fs2[(ky*3+kx)][c] * h[b,yy,xx,c] + bias(b)
// ---------------------------------------------------------------------------
__global__ __launch_bounds__(128)
void final_conv_kernel(const _Float16* __restrict__ h, const float* __restrict__ F,
                       float* __restrict__ out, int H, int W) {
    __shared__ float Fs2[9 * 256 + 1];   // [kk][c] transposed filter + bias
    int b = blockIdx.z, y = blockIdx.y;
    for (int i = threadIdx.x; i < 2304; i += 128) {
        int c = i / 9, kk = i % 9;
        Fs2[kk * 256 + c] = F[(size_t)b * ODIM + i];
    }
    if (threadIdx.x == 0) Fs2[2304] = F[(size_t)b * ODIM + 2304];
    __syncthreads();
    int x = blockIdx.x * 128 + threadIdx.x;
    if (x >= W) return;
    float acc = Fs2[2304];
#pragma unroll
    for (int ky = 0; ky < 3; ++ky) {
        int yy = y + ky - 1;
        if (yy < 0 || yy >= H) continue;
#pragma unroll
        for (int kx = 0; kx < 3; ++kx) {
            int xx = x + kx - 1;
            if (xx < 0 || xx >= W) continue;
            const _Float16* hp = h + (((size_t)b * H + yy) * W + xx) * 256;
            const float* fk = Fs2 + (ky * 3 + kx) * 256;
            for (int c = 0; c < 256; ++c)
                acc += fk[c] * (float)hp[c];
        }
    }
    out[((size_t)b * H + y) * W + x] = acc;
}

// ---------------------------------------------------------------------------
extern "C" void kernel_launch(void* const* d_in, const int* in_sizes, int n_in,
                              void* d_out, int out_size, void* d_ws, size_t ws_size,
                              hipStream_t stream) {
    (void)in_sizes; (void)n_in; (void)out_size; (void)ws_size;
    const float* x      = (const float*)d_in[0];
    const float* word   = (const float*)d_in[1];
    const float* score  = (const float*)d_in[2];
    const float* txt_w  = (const float*)d_in[3];
    const float* txt_b  = (const float*)d_in[4];
    const float* conv1w = (const float*)d_in[5];
    const float* bn1g = (const float*)d_in[6],  *bn1b = (const float*)d_in[7];
    const float* bn1m = (const float*)d_in[8],  *bn1v = (const float*)d_in[9];
    const float* conv2w = (const float*)d_in[10];
    const float* bn2g = (const float*)d_in[11], *bn2b = (const float*)d_in[12];
    const float* bn2m = (const float*)d_in[13], *bn2v = (const float*)d_in[14];
    const float* conv3w = (const float*)d_in[15];
    const float* conv3b = (const float*)d_in[16];
    float* out = (float*)d_out;

    const int B = BATCH, H1 = 52, W1 = 52, H2 = 104, W2 = 104;

    char* ws = (char*)d_ws;
    const size_t szX2 = (size_t)B * 512 * H1 * W1 * 2;   // 44,302,336
    const size_t szX4 = (size_t)B * 512 * H2 * W2 * 2;   // 177,209,344
    size_t oX2 = 0;
    size_t oC1 = oX2 + szX2;
    size_t oX4 = oC1 + szX2;
    size_t oW1 = oX4 + szX4;
    size_t oW2 = oW1 + (size_t)512 * 4608 * 2;
    size_t oW3 = oW2 + (size_t)256 * 4608 * 2;
    size_t oST = oW3 + (size_t)256 * 256 * 2;
    size_t oWB = oST + 2048 * 4;
    size_t oSB = oWB + (size_t)BATCH * WDIM * 4;
    size_t oF  = oSB + 256;
    size_t oC2 = oX2;   // reuse: x2+c1 region after conv1 is consumed
    size_t oH  = oX4;   // reuse: x4 region after conv2 is consumed

    _Float16* x2  = (_Float16*)(ws + oX2);   // NHWC (B,52,52,512)
    _Float16* c1o = (_Float16*)(ws + oC1);   // NHWC (B,52,52,512)
    _Float16* x4  = (_Float16*)(ws + oX4);   // NHWC (B,104,104,512)
    _Float16* c2o = (_Float16*)(ws + oC2);   // NHWC (B,104,104,256)
    _Float16* hbf = (_Float16*)(ws + oH);    // NHWC (B,104,104,256)
    _Float16* w1p = (_Float16*)(ws + oW1);
    _Float16* w2p = (_Float16*)(ws + oW2);
    _Float16* w3p = (_Float16*)(ws + oW3);
    float* st   = (float*)(ws + oST);
    float* wbar = (float*)(ws + oWB);
    float* sbar = (float*)(ws + oSB);
    float* Fbuf = (float*)(ws + oF);

    // constants prep
    prep_st_kernel<<<4, 256, 0, stream>>>(bn1g, bn1b, bn1m, bn1v,
                                          bn2g, bn2b, bn2m, bn2v, conv3b, st);
    pack_w_kernel<512, 3><<<(512 * 4608) / 256, 256, 0, stream>>>(conv1w, w1p, 512);
    pack_w_kernel<512, 3><<<(256 * 4608) / 256, 256, 0, stream>>>(conv2w, w2p, 256);
    pack_w_kernel<256, 1><<<(256 * 256) / 256, 256, 0, stream>>>(conv3w, w3p, 256);

    // vis path (all intermediates NHWC f16)
    {
        long long tot = (long long)B * H1 * W1 * 512;
        up2_kernel<float, true><<<(unsigned)((tot + 255) / 256), 256, 0, stream>>>(
            x, x2, B, 512, 26, 26);
    }
    conv_wmma_kernel<512, 3><<<dim3((W1 + 15) / 16, H1, B * (512 / 128)), 256, 0, stream>>>(
        x2, w1p, st, st + 512, c1o, H1, W1, 512, 1);
    {
        long long tot = (long long)B * H2 * W2 * 512;
        up2_kernel<_Float16, false><<<(unsigned)((tot + 255) / 256), 256, 0, stream>>>(
            c1o, x4, B, 512, H1, W1);
    }
    conv_wmma_kernel<512, 3><<<dim3((W2 + 15) / 16, H2, B * (256 / 128)), 256, 0, stream>>>(
        x4, w2p, st + 1024, st + 1280, c2o, H2, W2, 256, 1);
    conv_wmma_kernel<256, 1><<<dim3((W2 + 15) / 16, H2, B * (256 / 128)), 256, 0, stream>>>(
        c2o, w3p, st + 1536, st + 1792, hbf, H2, W2, 256, 0);

    // txt path (score-folded dynamic filter)
    fold_word_kernel<<<(BATCH * WDIM) / 256, 256, 0, stream>>>(word, score, wbar, sbar);
    dyn_filter_kernel<<<(BATCH * ODIM + 255) / 256, 256, 0, stream>>>(txt_w, txt_b, wbar, sbar, Fbuf);

    // folded dynamic grouped conv -> output (B,1,104,104) f32
    final_conv_kernel<<<dim3((W2 + 127) / 128, H2, B), 128, 0, stream>>>(hbf, Fbuf, out, H2, W2);
}
